// bwd_gnn_45174466019866
// MI455X (gfx1250) — matrix operand order repaired and checked by reading, not compiled, not run on hardware
//
#include <hip/hip_runtime.h>

#define HDIM   256
#define FDIM   128
#define PNODES 5000
#define NLEV   20
#define EPL    40000
#define NNODES (NLEV * PNODES)

typedef __bf16 bf16;
typedef __attribute__((ext_vector_type(16))) __bf16 v16bf;
typedef __attribute__((ext_vector_type(8)))  __bf16 v8bf;
typedef __attribute__((ext_vector_type(8)))  float  v8f;

#define BM 128
#define BN 64
#define BK 32
#define LPAD 8          // LDS row stride BK+8 = 40 elems = 80B (16B aligned, bank-conflict-free)

// native fptrunc -> hardware bf16 convert on gfx1250
__device__ __forceinline__ bf16 to_bf16(float f) { return (bf16)f; }

// out[M, HDIM] = act( A @ W + bias )
// A assembled from f32 sources; Wt is bf16, PRE-TRANSPOSED: Wt[n*Ktot + k] = W[k, n]
// MODE 0: A = A0 (K0 cols) | MODE 1: A = A0 + A1 | MODE 2: A = concat(A0[K0], A1[K1])
// ACT 0: relu | ACT 1: tanh
template<int MODE, int ACT>
__global__ __launch_bounds__(256)
void gemm_wmma_bf16(const float* __restrict__ A0, const float* __restrict__ A1,
                    const bf16* __restrict__ Wt, const float* __restrict__ bias,
                    float* __restrict__ out, int M, int K0, int K1)
{
    const int Ktot = (MODE == 2) ? (K0 + K1) : K0;

    __shared__ __align__(16) bf16 As[2][BM][BK + LPAD];
    __shared__ __align__(16) bf16 Bs[2][BN][BK + LPAD];   // B stored transposed: [n][k]

    const int tid  = threadIdx.x;
    const int lane = tid & 31;
    const int wave = tid >> 5;                 // 0..7, one 16-row strip each
    const int bm0  = blockIdx.x * BM;
    const int bn0  = blockIdx.y * BN;

    v8f acc[4];
    #pragma unroll
    for (int t = 0; t < 4; ++t) acc[t] = (v8f){0.f,0.f,0.f,0.f,0.f,0.f,0.f,0.f};

    // staging assignments
    const int arow  = tid >> 1;                // 0..127
    const int acol0 = (tid & 1) * 16;          // 0 or 16 (16 floats = 64B contiguous)
    const int brn   = tid >> 2;                // 0..63 (output column)
    const int bk0   = (tid & 3) * 8;           // 0,8,16,24 (8 bf16 = 16B contiguous)
    const int arg   = bm0 + arow;

    float4 ra[4];
    v8bf   rb;

    auto fetch = [&](int kb) {
        const int kg0 = kb + acol0;
        if (arg < M) {
            if (MODE == 0) {
                const float* p = A0 + (size_t)arg * K0 + kg0;
                #pragma unroll
                for (int i = 0; i < 4; ++i) ra[i] = *(const float4*)(p + 4 * i);
            } else if (MODE == 1) {
                const float* p0 = A0 + (size_t)arg * K0 + kg0;
                const float* p1 = A1 + (size_t)arg * K0 + kg0;
                #pragma unroll
                for (int i = 0; i < 4; ++i) {
                    float4 u = *(const float4*)(p0 + 4 * i);
                    float4 w = *(const float4*)(p1 + 4 * i);
                    ra[i] = make_float4(u.x + w.x, u.y + w.y, u.z + w.z, u.w + w.w);
                }
            } else {
                const float* p = (kg0 < K0) ? (A0 + (size_t)arg * K0 + kg0)
                                            : (A1 + (size_t)arg * K1 + (kg0 - K0));
                #pragma unroll
                for (int i = 0; i < 4; ++i) ra[i] = *(const float4*)(p + 4 * i);
            }
        } else {
            #pragma unroll
            for (int i = 0; i < 4; ++i) ra[i] = make_float4(0.f, 0.f, 0.f, 0.f);
        }
        rb = *(const v8bf*)(Wt + (size_t)(bn0 + brn) * Ktot + kb + bk0);
    };

    auto stage = [&](int buf) {
        const float* af = (const float*)ra;
        v8bf p0, p1;
        #pragma unroll
        for (int j = 0; j < 8; ++j) p0[j] = to_bf16(af[j]);
        #pragma unroll
        for (int j = 0; j < 8; ++j) p1[j] = to_bf16(af[8 + j]);
        *(v8bf*)&As[buf][arow][acol0]     = p0;
        *(v8bf*)&As[buf][arow][acol0 + 8] = p1;
        *(v8bf*)&Bs[buf][brn][bk0]        = rb;
    };

    fetch(0);
    stage(0);
    __syncthreads();

    const int nk = Ktot / BK;
    for (int ki = 0; ki < nk; ++ki) {
        const int  buf  = ki & 1;
        const bool more = (ki + 1 < nk);
        if (more) fetch((ki + 1) * BK);

        // ---- A fragment (16x32 bf16): two 16B DS loads per lane ----
        const int m   = lane & 15;
        const int kk8 = (lane >> 4) * 8;       // lanes 16..31: K base 8 / 24
        v8bf alo = *(const v8bf*)&As[buf][wave * 16 + m][kk8];
        v8bf ahi = *(const v8bf*)&As[buf][wave * 16 + m][kk8 + 16];
        v16bf afrag = __builtin_shufflevector(alo, ahi,
            0,1,2,3,4,5,6,7,8,9,10,11,12,13,14,15);

        // ---- load ALL 4 B fragments first (batch DS, single wait) ----
        const int n    = lane & 15;
        const int kk16 = (lane >> 4) * 8;      // lanes 16..31: K base 8 / 24
        v16bf bfrag[4];
        #pragma unroll
        for (int ct = 0; ct < 4; ++ct) {
            v8bf blo = *(const v8bf*)&Bs[buf][ct * 16 + n][kk16];
            v8bf bhi = *(const v8bf*)&Bs[buf][ct * 16 + n][kk16 + 16];
            bfrag[ct] = __builtin_shufflevector(blo, bhi,
                0,1,2,3,4,5,6,7,8,9,10,11,12,13,14,15);
        }
        // ---- 4 independent WMMAs back-to-back ----
        #pragma unroll
        for (int ct = 0; ct < 4; ++ct) {
            acc[ct] = __builtin_amdgcn_wmma_f32_16x16x32_bf16(
                false, afrag, false, bfrag[ct], (short)0, acc[ct], false, false);
        }

        if (more) stage(buf ^ 1);
        __syncthreads();
    }

    // ---- epilogue: bias + activation + store (f32 C layout: VGPR r -> M = r + 8*(lane>=16)) ----
    const int n_local = lane & 15;
    const int rbase   = (lane >> 4) * 8;
    #pragma unroll
    for (int ct = 0; ct < 4; ++ct) {
        const int cg = bn0 + ct * 16 + n_local;
        const float bv = bias[cg];
        #pragma unroll
        for (int r = 0; r < 8; ++r) {
            const int rg = bm0 + wave * 16 + rbase + r;
            if (rg < M) {
                float v = acc[ct][r] + bv;
                v = (ACT == 0) ? fmaxf(v, 0.f) : tanhf(v);
                out[(size_t)rg * HDIM + cg] = v;
            }
        }
    }
}

__global__ void zero_f32(float* __restrict__ p, int n) {
    int i = blockIdx.x * blockDim.x + threadIdx.x;
    if (i < n) p[i] = 0.f;
}

// Wt[n*K + k] = bf16( W[k*N + n] )   (transpose + convert, one-time)
__global__ void f32_to_bf16_T(const float* __restrict__ in, bf16* __restrict__ out,
                              int K, int N) {
    int i = blockIdx.x * blockDim.x + threadIdx.x;     // over K*N outputs
    if (i >= K * N) return;
    int k = i % K, n = i / K;
    out[(size_t)n * K + k] = to_bf16(in[(size_t)k * N + n]);
}

// mailbox: msgs[dst] += embeds[src] (one thread = one edge x 4 consecutive floats)
__global__ __launch_bounds__(256)
void scatter_add(const float* __restrict__ embeds, const int* __restrict__ src,
                 const int* __restrict__ dst, float* __restrict__ msgs)
{
    int t  = blockIdx.x * blockDim.x + threadIdx.x;
    int e  = t >> 6;               // HDIM/4 = 64 chunks per edge
    int c4 = (t & 63) * 4;
    if (e >= EPL) return;
    const int s = src[e];
    const int d = dst[e];
    const float4 v = *(const float4*)(embeds + (size_t)s * HDIM + c4);
    float* mp = msgs + (size_t)d * HDIM + c4;
    atomicAdd(mp + 0, v.x);
    atomicAdd(mp + 1, v.y);
    atomicAdd(mp + 2, v.z);
    atomicAdd(mp + 3, v.w);
}

extern "C" void kernel_launch(void* const* d_in, const int* in_sizes, int n_in,
                              void* d_out, int out_size, void* d_ws, size_t ws_size,
                              hipStream_t stream) {
    (void)in_sizes; (void)n_in; (void)out_size; (void)ws_size;

    const float* node_feats = (const float*)d_in[0];   // [N, 128]
    const float* W_embed    = (const float*)d_in[1];   // [128, 256]
    const float* b_embed    = (const float*)d_in[2];   // [256]
    const float* mp_w       = (const float*)d_in[3];   // [4, 256, 256]
    const float* mp_b       = (const float*)d_in[4];   // [4, 256]
    const float* ne_w0      = (const float*)d_in[5];   // [512, 256]
    const float* ne_b0      = (const float*)d_in[6];   // [256]
    const float* ne_w       = (const float*)d_in[7];   // [3, 256, 256]
    const float* ne_b       = (const float*)d_in[8];   // [3, 256]
    const int*   edge_src   = (const int*)d_in[9];     // [19, 40000] global ids
    const int*   edge_dst   = (const int*)d_in[10];    // [19, 40000] local ids

    float* embeds = (float*)d_out;                     // [N, 256], doubles as base

    char*  ws  = (char*)d_ws;
    size_t off = 0;
    auto carve = [&](size_t bytes) -> void* {
        void* p = ws + off;
        off = (off + bytes + 255) & ~(size_t)255;
        return p;
    };

    const int HH = HDIM * HDIM;
    bf16* wt_embed = (bf16*)carve((size_t)FDIM * HDIM * sizeof(bf16));  // [256][128]
    bf16* wt_mp    = (bf16*)carve((size_t)4 * HH * sizeof(bf16));       // 4x [256][256]
    bf16* wt_ne0   = (bf16*)carve((size_t)2 * HH * sizeof(bf16));       // [256][512]
    bf16* wt_ne    = (bf16*)carve((size_t)3 * HH * sizeof(bf16));       // 3x [256][256]

    const size_t actB = (size_t)PNODES * HDIM * sizeof(float);
    float* msgs = (float*)carve(actB);
    float* t0   = (float*)carve(actB);
    float* t1   = (float*)carve(actB);
    float* t2   = (float*)carve(actB);
    float* mr   = (float*)carve(actB);
    float* x0   = (float*)carve(actB);
    float* x1   = (float*)carve(actB);
    float* x2   = (float*)carve(actB);

    // ---- one-time weight transpose + bf16 conversion ----
    auto cvtT = [&](const float* src, bf16* dst, int K, int N) {
        int n = K * N;
        f32_to_bf16_T<<<(n + 255) / 256, 256, 0, stream>>>(src, dst, K, N);
    };
    cvtT(W_embed, wt_embed, FDIM, HDIM);
    for (int i = 0; i < 4; ++i) cvtT(mp_w + (size_t)i * HH, wt_mp + (size_t)i * HH, HDIM, HDIM);
    cvtT(ne_w0, wt_ne0, 2 * HDIM, HDIM);
    for (int i = 0; i < 3; ++i) cvtT(ne_w + (size_t)i * HH, wt_ne + (size_t)i * HH, HDIM, HDIM);

    // ---- embed: base = tanh(node_feats @ W_embed + b) for all N nodes ----
    {
        dim3 grid((NNODES + BM - 1) / BM, HDIM / BN);
        gemm_wmma_bf16<0, 1><<<grid, 256, 0, stream>>>(
            node_feats, nullptr, wt_embed, b_embed, embeds, NNODES, FDIM, 0);
    }

    const dim3 gridP((PNODES + BM - 1) / BM, HDIM / BN);
    const int nMsg = PNODES * HDIM;

    for (int l = 0; l < NLEV - 1; ++l) {
        // mailbox: segment-sum of level-l embeds into level l+1 slots
        zero_f32<<<(nMsg + 255) / 256, 256, 0, stream>>>(msgs, nMsg);
        scatter_add<<<(EPL * 64 + 255) / 256, 256, 0, stream>>>(
            embeds, edge_src + (size_t)l * EPL, edge_dst + (size_t)l * EPL, msgs);

        // redux_func: mp residual stack
        gemm_wmma_bf16<0, 0><<<gridP, 256, 0, stream>>>(
            msgs, nullptr, wt_mp + 0 * HH, mp_b + 0 * HDIM, t0, PNODES, HDIM, 0);
        gemm_wmma_bf16<0, 0><<<gridP, 256, 0, stream>>>(
            t0, nullptr, wt_mp + 1 * HH, mp_b + 1 * HDIM, t1, PNODES, HDIM, 0);
        gemm_wmma_bf16<1, 0><<<gridP, 256, 0, stream>>>(
            t0, t1, wt_mp + 2 * HH, mp_b + 2 * HDIM, t2, PNODES, HDIM, 0);
        gemm_wmma_bf16<1, 0><<<gridP, 256, 0, stream>>>(
            t1, t2, wt_mp + 3 * HH, mp_b + 3 * HDIM, mr, PNODES, HDIM, 0);

        // apply_func: concat(base_slice, mr) then ne residual stack
        float* base_slice = embeds + (size_t)(l + 1) * PNODES * HDIM;
        gemm_wmma_bf16<2, 0><<<gridP, 256, 0, stream>>>(
            base_slice, mr, wt_ne0, ne_b0, x0, PNODES, HDIM, HDIM);
        gemm_wmma_bf16<0, 0><<<gridP, 256, 0, stream>>>(
            x0, nullptr, wt_ne + 0 * HH, ne_b + 0 * HDIM, x1, PNODES, HDIM, 0);
        gemm_wmma_bf16<1, 0><<<gridP, 256, 0, stream>>>(
            x0, x1, wt_ne + 1 * HH, ne_b + 1 * HDIM, x2, PNODES, HDIM, 0);
        gemm_wmma_bf16<1, 0><<<gridP, 256, 0, stream>>>(
            x1, x2, wt_ne + 2 * HH, ne_b + 2 * HDIM, base_slice, PNODES, HDIM, 0);
    }
}
